// MAB_25503515803817
// MI455X (gfx1250) — compile-verified, matrix operand order checked by reading, not run on hardware
//
#include <hip/hip_runtime.h>
#include <hip/hip_bf16.h>
#include <math.h>

#define BB 8
#define LL 1024
#define DV 512
#define HH 8
#define DH 64
#define EPSV 1e-5f

typedef __attribute__((ext_vector_type(8)))  __bf16 v8bf;
typedef __attribute__((ext_vector_type(16))) __bf16 v16bf;
typedef __attribute__((ext_vector_type(8)))  float  v8f;

__device__ __forceinline__ unsigned short f2bf(float f) {
    union { float f; unsigned int u; } c; c.f = f;
    unsigned int u = c.u;
    u += 0x7FFFu + ((u >> 16) & 1u);            // round-to-nearest-even
    return (unsigned short)(u >> 16);
}
__device__ __forceinline__ float bf2f(unsigned short h) {
    union { unsigned int u; float f; } c; c.u = ((unsigned int)h) << 16;
    return c.f;
}

// Load one 16x32 bf16 A/B fragment per the CDNA5 WMMA VGPR layout:
// lane L: row = base + (L&15); elements 0..7 = K {0..7}+8*(L>>4),
// elements 8..15 = K {16..23}+8*(L>>4). Two 16B vector loads per lane.
__device__ __forceinline__ v16bf load_frag(const unsigned short* base, int ld,
                                           int row, int k0, int half) {
    const __bf16* p = (const __bf16*)base + (long)row * ld + k0 + 8 * half;
    v8bf lo = *(const v8bf*)p;
    v8bf hi = *(const v8bf*)(p + 16);
    v16bf r;
#pragma unroll
    for (int i = 0; i < 8; i++) { r[i] = lo[i]; r[i + 8] = hi[i]; }
    return r;
}

// Generic batched C[M,N] = alpha * A[M,K] * B[N,K]^T + bias (+ residual).
// Each wave computes a 64x64 tile: 16 f32 accumulators, 8 fragment loads and
// 16 v_wmma_f32_16x16x32_bf16 per 32-deep K step (~16 flops/byte from cache).
// M and N must be multiples of 64. Batch via blockIdx.y/z strides.
__global__ __launch_bounds__(256)
void wmma_gemm_bf16(const unsigned short* __restrict__ A, long sAy, long sAz, int lda,
                    const unsigned short* __restrict__ Bm, long sBy, long sBz, int ldb,
                    float* Cf, unsigned short* Cbf, long sCy, long sCz, int ldc,
                    const float* __restrict__ bias, const float* __restrict__ Res,
                    int M, int N, int K, float alpha) {
    int ntn  = N >> 6;
    int wave = (int)((blockIdx.x * blockDim.x + threadIdx.x) >> 5);
    if (wave >= (M >> 6) * ntn) return;          // whole-wave exit: EXEC stays all-1s
    int lane = threadIdx.x & 31;
    int tm = (wave / ntn) << 6;
    int tn = (wave % ntn) << 6;

    const unsigned short* Ab = A  + blockIdx.y * sAy + blockIdx.z * sAz;
    const unsigned short* Bb = Bm + blockIdx.y * sBy + blockIdx.z * sBz;
    long cofs = blockIdx.y * sCy + blockIdx.z * sCz;

    int row = lane & 15, half = lane >> 4;
    v8f acc[4][4] = {};

    for (int k0 = 0; k0 < K; k0 += 32) {
        v16bf a[4], b[4];
#pragma unroll
        for (int i = 0; i < 4; i++) a[i] = load_frag(Ab, lda, tm + 16 * i + row, k0, half);
#pragma unroll
        for (int j = 0; j < 4; j++) b[j] = load_frag(Bb, ldb, tn + 16 * j + row, k0, half);
#pragma unroll
        for (int i = 0; i < 4; i++)
#pragma unroll
            for (int j = 0; j < 4; j++)
                acc[i][j] = __builtin_amdgcn_wmma_f32_16x16x32_bf16(
                    false, a[i], false, b[j], (short)0, acc[i][j], false, false);
    }

    // C/D layout: vgpr r -> row (lane>>4)*8 + r, col lane&15 within 16x16 tile
    int nin = lane & 15, mo = (lane >> 4) * 8;
#pragma unroll
    for (int i = 0; i < 4; i++) {
#pragma unroll
        for (int j = 0; j < 4; j++) {
            int gn = tn + j * 16 + nin;
            float bv = bias ? bias[gn] : 0.0f;
#pragma unroll
            for (int r = 0; r < 8; r++) {
                int gm = tm + i * 16 + mo + r;
                long idx = cofs + (long)gm * ldc + gn;
                float v = acc[i][j][r] * alpha + bv;
                if (Res) v += Res[idx];
                if (Cf)  Cf[idx] = v;
                if (Cbf) Cbf[idx] = f2bf(v);
            }
        }
    }
}

__global__ void f2bf_kernel(const float* __restrict__ src, unsigned short* __restrict__ dst, long n) {
    long i = (long)blockIdx.x * blockDim.x + threadIdx.x;
    long stride = (long)gridDim.x * blockDim.x;
    for (; i < n; i += stride) dst[i] = f2bf(src[i]);
}

// Vt[((h*BB+b)*DH + d)*LL + k] = Vp[(b*LL + k)*DV + h*DH + d]
__global__ void vtrans_kernel(const unsigned short* __restrict__ Vp, unsigned short* __restrict__ Vt) {
    long n = (long)HH * BB * DH * LL;
    long i = (long)blockIdx.x * blockDim.x + threadIdx.x;
    long stride = (long)gridDim.x * blockDim.x;
    for (; i < n; i += stride) {
        int k = (int)(i & (LL - 1));
        long t = i >> 10;
        int d = (int)(t & (DH - 1)); t >>= 6;
        int b = (int)(t & (BB - 1));
        int h = (int)(t >> 3);
        Vt[i] = Vp[((long)b * LL + k) * DV + h * DH + d];
    }
}

// Column softmax over the QUERY axis of S[h,b,q,k] (in place, bf16).
__global__ void softmax_q_kernel(unsigned short* __restrict__ S) {
    int k = blockIdx.x * 256 + threadIdx.x;                 // 0..LL-1
    long base = ((long)blockIdx.z * BB + blockIdx.y) * (long)LL * LL;
    unsigned short* col = S + base + k;
    float m = -1e30f;
    for (int q = 0; q < LL; q++) m = fmaxf(m, bf2f(col[(long)q * LL]));
    float s = 0.0f;
    for (int q = 0; q < LL; q++) s += __expf(bf2f(col[(long)q * LL]) - m);
    float inv = 1.0f / s;
    for (int q = 0; q < LL; q++) {
        long idx = (long)q * LL;
        col[idx] = f2bf(__expf(bf2f(col[idx]) - m) * inv);
    }
}

// LayerNorm(512) per row -> f32 out + bf16 copy (for the next WMMA GEMM).
__global__ __launch_bounds__(256)
void ln0_kernel(const float* __restrict__ X, const float* __restrict__ g,
                const float* __restrict__ bta, float* __restrict__ Yf,
                unsigned short* __restrict__ Ybf) {
    long row = blockIdx.x;
    const float* x = X + row * DV;
    int tid = threadIdx.x;
    float v0 = x[tid], v1 = x[tid + 256];
    __shared__ float s1[256], s2[256];
    s1[tid] = v0 + v1;
    s2[tid] = v0 * v0 + v1 * v1;
    __syncthreads();
    for (int off = 128; off > 0; off >>= 1) {
        if (tid < off) { s1[tid] += s1[tid + off]; s2[tid] += s2[tid + off]; }
        __syncthreads();
    }
    float mean = s1[0] * (1.0f / DV);
    float var  = s2[0] * (1.0f / DV) - mean * mean;
    float rs   = rsqrtf(var + EPSV);
    float y0 = (v0 - mean) * rs * g[tid]       + bta[tid];
    float y1 = (v1 - mean) * rs * g[tid + 256] + bta[tid + 256];
    Yf[row * DV + tid]        = y0;  Ybf[row * DV + tid]        = f2bf(y0);
    Yf[row * DV + tid + 256]  = y1;  Ybf[row * DV + tid + 256]  = f2bf(y1);
}

// t = y + relu(r); out = LN(t) with g1/b1.
__global__ __launch_bounds__(256)
void final_kernel(const float* __restrict__ Y, const float* __restrict__ R,
                  const float* __restrict__ g, const float* __restrict__ bta,
                  float* __restrict__ out) {
    long row = blockIdx.x;
    int tid = threadIdx.x;
    float t0 = Y[row * DV + tid]       + fmaxf(R[row * DV + tid], 0.0f);
    float t1 = Y[row * DV + tid + 256] + fmaxf(R[row * DV + tid + 256], 0.0f);
    __shared__ float s1[256], s2[256];
    s1[tid] = t0 + t1;
    s2[tid] = t0 * t0 + t1 * t1;
    __syncthreads();
    for (int off = 128; off > 0; off >>= 1) {
        if (tid < off) { s1[tid] += s1[tid + off]; s2[tid] += s2[tid + off]; }
        __syncthreads();
    }
    float mean = s1[0] * (1.0f / DV);
    float var  = s2[0] * (1.0f / DV) - mean * mean;
    float rs   = rsqrtf(var + EPSV);
    out[row * DV + tid]       = (t0 - mean) * rs * g[tid]       + bta[tid];
    out[row * DV + tid + 256] = (t1 - mean) * rs * g[tid + 256] + bta[tid + 256];
}

static inline dim3 gemm_grid(int M, int N, int ny, int nz) {
    int waves = (M >> 6) * (N >> 6);
    return dim3((unsigned)((waves * 32 + 255) / 256), (unsigned)ny, (unsigned)nz);
}

extern "C" void kernel_launch(void* const* d_in, const int* in_sizes, int n_in,
                              void* d_out, int out_size, void* d_ws, size_t ws_size,
                              hipStream_t stream) {
    (void)in_sizes; (void)n_in; (void)out_size; (void)ws_size;
    const float* Q  = (const float*)d_in[0];
    const float* K  = (const float*)d_in[1];
    const float* Wq = (const float*)d_in[2];
    const float* bq = (const float*)d_in[3];
    const float* Wk = (const float*)d_in[4];
    const float* bk = (const float*)d_in[5];
    const float* Wv = (const float*)d_in[6];
    const float* bv = (const float*)d_in[7];
    const float* Wo = (const float*)d_in[8];
    const float* bo = (const float*)d_in[9];
    const float* g0 = (const float*)d_in[10];
    const float* b0 = (const float*)d_in[11];
    const float* g1 = (const float*)d_in[12];
    const float* b1 = (const float*)d_in[13];
    float* out = (float*)d_out;

    const long NROW = (long)BB * LL;              // 8192
    const long NTOK = NROW * DV;                  // 4,194,304
    const long NSCORE = (long)HH * BB * LL * LL;  // 67,108,864

    // --- workspace carve-up (256B aligned) ---
    size_t off = 0;
    char* base = (char*)d_ws;
    auto alloc = [&](size_t bytes) -> char* {
        char* p = base + off;
        off += (bytes + 255) & ~(size_t)255;
        return p;
    };
    unsigned short* qbf   = (unsigned short*)alloc(NTOK * 2);
    unsigned short* kbf   = (unsigned short*)alloc(NTOK * 2);
    unsigned short* wq_bf = (unsigned short*)alloc((size_t)DV * DV * 2);
    unsigned short* wk_bf = (unsigned short*)alloc((size_t)DV * DV * 2);
    unsigned short* wv_bf = (unsigned short*)alloc((size_t)DV * DV * 2);
    unsigned short* wo_bf = (unsigned short*)alloc((size_t)DV * DV * 2);
    float*          qp_f  = (float*)alloc(NTOK * 4);
    unsigned short* qp_bf = (unsigned short*)alloc(NTOK * 2);
    unsigned short* kp_bf = (unsigned short*)alloc(NTOK * 2);
    unsigned short* vp_bf = (unsigned short*)alloc(NTOK * 2);
    unsigned short* vt_bf = (unsigned short*)alloc(NTOK * 2);
    unsigned short* Sbuf  = (unsigned short*)alloc(NSCORE * 2);
    float*          Xbuf  = (float*)alloc(NTOK * 4);
    float*          Ybuf  = (float*)alloc(NTOK * 4);
    unsigned short* ybf   = (unsigned short*)alloc(NTOK * 2);
    float*          Rbuf  = (float*)alloc(NTOK * 4);

    // --- 1) converts to bf16 ---
    f2bf_kernel<<<2048, 256, 0, stream>>>(Q,  qbf, NTOK);
    f2bf_kernel<<<2048, 256, 0, stream>>>(K,  kbf, NTOK);
    f2bf_kernel<<<512,  256, 0, stream>>>(Wq, wq_bf, (long)DV * DV);
    f2bf_kernel<<<512,  256, 0, stream>>>(Wk, wk_bf, (long)DV * DV);
    f2bf_kernel<<<512,  256, 0, stream>>>(Wv, wv_bf, (long)DV * DV);
    f2bf_kernel<<<512,  256, 0, stream>>>(Wo, wo_bf, (long)DV * DV);

    // --- 2) projections: [8192,512] x [512,512]^T ---
    wmma_gemm_bf16<<<gemm_grid(NROW, DV, 1, 1), 256, 0, stream>>>(
        qbf, 0, 0, DV,  wq_bf, 0, 0, DV,
        qp_f, qp_bf, 0, 0, DV, bq, nullptr, (int)NROW, DV, DV, 1.0f);
    wmma_gemm_bf16<<<gemm_grid(NROW, DV, 1, 1), 256, 0, stream>>>(
        kbf, 0, 0, DV,  wk_bf, 0, 0, DV,
        nullptr, kp_bf, 0, 0, DV, bk, nullptr, (int)NROW, DV, DV, 1.0f);
    wmma_gemm_bf16<<<gemm_grid(NROW, DV, 1, 1), 256, 0, stream>>>(
        kbf, 0, 0, DV,  wv_bf, 0, 0, DV,
        nullptr, vp_bf, 0, 0, DV, bv, nullptr, (int)NROW, DV, DV, 1.0f);

    // --- 3) V transpose to [h,b,d,k] for the AV GEMM ---
    vtrans_kernel<<<4096, 256, 0, stream>>>(vp_bf, vt_bf);

    // --- 4) scores: per (h,b) [1024,64] x [1024,64]^T, alpha = 1/sqrt(512) ---
    const float alpha = 1.0f / sqrtf((float)DV);
    wmma_gemm_bf16<<<gemm_grid(LL, LL, BB, HH), 256, 0, stream>>>(
        qp_bf, (long)LL * DV, (long)DH, DV,
        kp_bf, (long)LL * DV, (long)DH, DV,
        nullptr, Sbuf, (long)LL * LL, (long)BB * LL * LL, LL,
        nullptr, nullptr, LL, LL, DH, alpha);

    // --- 5) softmax over the query axis (column-wise per (h,b)) ---
    softmax_q_kernel<<<dim3(LL / 256, BB, HH), 256, 0, stream>>>(Sbuf);

    // --- 6) O = A @ V, with Qp residual fused: X[b,q,h*64+d] = Qp + O ---
    wmma_gemm_bf16<<<gemm_grid(LL, DH, BB, HH), 256, 0, stream>>>(
        Sbuf,  (long)LL * LL, (long)BB * LL * LL, LL,
        vt_bf, (long)DH * LL, (long)BB * DH * LL, LL,
        Xbuf, nullptr, (long)LL * DV, (long)DH, DV,
        nullptr, qp_f, LL, DH, LL, 1.0f);

    // --- 7) LN0 ---
    ln0_kernel<<<(unsigned)NROW, 256, 0, stream>>>(Xbuf, g0, b0, Ybuf, ybf);

    // --- 8) Wo GEMM: R = Y @ Wo^T + bo ---
    wmma_gemm_bf16<<<gemm_grid(NROW, DV, 1, 1), 256, 0, stream>>>(
        ybf, 0, 0, DV,  wo_bf, 0, 0, DV,
        Rbuf, nullptr, 0, 0, DV, bo, nullptr, (int)NROW, DV, DV, 1.0f);

    // --- 9) out = LN1(Y + relu(R)) ---
    final_kernel<<<(unsigned)NROW, 256, 0, stream>>>(Ybuf, Rbuf, g1, b1, out);
}
